// YOLOv3WithResNetBackbone_18837726560581
// MI455X (gfx1250) — compile-verified, matrix operand order checked by reading, not run on hardware
//
#include <hip/hip_runtime.h>
#include <float.h>

typedef __attribute__((ext_vector_type(16))) _Float16 v16h;
typedef __attribute__((ext_vector_type(8)))  float    v8f;

#define NUM_B 16
#define NUM_A 3
#define NUM_C 20
#define ATTRS 25
#define GH 52
#define GW 52
#define HW 2704          // 52*52
#define PRED 8112        // NUM_A*HW
#define PADN 8192        // next pow2 for bitonic sort
#define TOPK 2048
#define STRIDE_F 8.0f    // 416/52
#define CONF_TH 0.5f
#define NMS_TH 0.4f
#define MASK_WORDS 64    // 2048/32

// ---- d_out layout (float elements) ----
#define OUT_DECODE 0
#define OUT_TOPIDX (NUM_B * PRED * ATTRS)        // int32 bits stored here
#define OUT_KEEP   (OUT_TOPIDX + NUM_B * TOPK)   // 0.0f / 1.0f

// ---- workspace layout (bytes) ----
#define WS_CONF    0u
#define WS_TOPCONF (WS_CONF    + NUM_B * PADN * 4u)
#define WS_TOPIDX  (WS_TOPCONF + NUM_B * TOPK * 4u)
#define WS_CORN    (WS_TOPIDX  + NUM_B * TOPK * 4u)
#define WS_CLS     (WS_CORN    + NUM_B * TOPK * 16u)
#define WS_MASK    (WS_CLS     + NUM_B * TOPK * 4u)

__device__ __forceinline__ float sigf(float x) { return 1.0f / (1.0f + __expf(-x)); }

// ---------------------------------------------------------------------------
// Kernel 1: decode. One block = 256 grid cells of one (b, anchor).
// Coalesced attr-plane reads, LDS transpose, contiguous [hw x 25] writes.
// ---------------------------------------------------------------------------
__global__ void __launch_bounds__(256)
decode_kernel(const float* __restrict__ in, const float* __restrict__ anchors,
              float* __restrict__ out, float* __restrict__ conf_ws)
{
    __shared__ float tile[256 * ATTRS];
    const int tid = threadIdx.x;
    const int hw0 = blockIdx.x * 256;
    const int a   = blockIdx.y;
    const int b   = blockIdx.z;
    const int hw  = hw0 + tid;
    const int n   = min(256, HW - hw0);
    const float aw = anchors[a * 2 + 0];
    const float ah = anchors[a * 2 + 1];

    if (tid < n) {
        const float gx = (float)(hw % GW);
        const float gy = (float)(hw / GW);
        const float* ip = in + (size_t)(b * NUM_A + a) * ATTRS * HW + hw;
        #pragma unroll
        for (int attr = 0; attr < ATTRS; ++attr) {
            float v = ip[(size_t)attr * HW];
            float r;
            if      (attr == 0) r = (sigf(v) + gx) * STRIDE_F;            // bx
            else if (attr == 1) r = (sigf(v) + gy) * STRIDE_F;            // by
            else if (attr == 2) r = __expf(v) * aw;                        // bw = exp(w)*anchor
            else if (attr == 3) r = __expf(v) * ah;                        // bh
            else                r = sigf(v);                               // conf + classes
            tile[tid * ATTRS + attr] = r;
            if (attr == 4) conf_ws[(size_t)b * PADN + a * HW + hw] = r;
        }
    }
    __syncthreads();
    const int total = n * ATTRS;
    float* op = out + ((size_t)b * PRED + a * HW + hw0) * ATTRS;
    for (int t = tid; t < total; t += 256) op[t] = tile[t];
}

// ---------------------------------------------------------------------------
// Kernel 2: full bitonic sort (desc) of 8192 (conf, idx) pairs per image in
// LDS; emit sorted top-2048 conf + indices.
// ---------------------------------------------------------------------------
__global__ void __launch_bounds__(1024)
topk_kernel(const float* __restrict__ conf_ws,
            float* __restrict__ topconf, int* __restrict__ topidx,
            int* __restrict__ out_topidx)
{
    __shared__ float keys[PADN];
    __shared__ int   vals[PADN];
    const int tid = threadIdx.x;
    const int b   = blockIdx.x;

    for (int t = tid; t < PADN; t += 1024) {
        keys[t] = (t < PRED) ? conf_ws[(size_t)b * PADN + t] : -FLT_MAX;
        vals[t] = t;
    }
    for (int k = 2; k <= PADN; k <<= 1) {
        for (int j = k >> 1; j > 0; j >>= 1) {
            __syncthreads();
            for (int p = tid; p < PADN / 2; p += 1024) {
                int i   = ((p & ~(j - 1)) << 1) | (p & (j - 1));
                int ixj = i | j;
                bool desc = ((i & k) == 0);
                float ka = keys[i], kb = keys[ixj];
                if (desc ? (ka < kb) : (ka > kb)) {
                    keys[i] = kb; keys[ixj] = ka;
                    int tv = vals[i]; vals[i] = vals[ixj]; vals[ixj] = tv;
                }
            }
        }
    }
    __syncthreads();
    for (int t = tid; t < TOPK; t += 1024) {
        topconf[b * TOPK + t]    = keys[t];
        topidx[b * TOPK + t]     = vals[t];
        out_topidx[b * TOPK + t] = vals[t];
    }
}

// ---------------------------------------------------------------------------
// Kernel 3: gather corners (x1y1x2y2) + argmax class for the top-K.
// ---------------------------------------------------------------------------
__global__ void __launch_bounds__(256)
gather_kernel(const float* __restrict__ out, const int* __restrict__ topidx,
              float4* __restrict__ corners, int* __restrict__ cls)
{
    int g = blockIdx.x * blockDim.x + threadIdx.x;
    if (g >= NUM_B * TOPK) return;
    int b   = g / TOPK;
    int idx = topidx[g];
    const float* p = out + ((size_t)b * PRED + idx) * ATTRS;
    __builtin_prefetch(p, 0, 1);   // global_prefetch_b8
    float cx = p[0], cy = p[1], w = p[2], h = p[3];
    float4 c;
    c.x = cx - w * 0.5f; c.y = cy - h * 0.5f;
    c.z = cx + w * 0.5f; c.w = cy + h * 0.5f;
    corners[g] = c;
    float best = p[5]; int bi = 0;
    #pragma unroll
    for (int ci = 1; ci < NUM_C; ++ci) {
        float v = p[5 + ci];
        if (v > best) { best = v; bi = ci; }    // strict '>' keeps first max (argmax semantics)
    }
    cls[g] = bi;
}

// ---------------------------------------------------------------------------
// Kernel 4: suppression bitmask. One wave computes a 16-row x 32-col tile of
// the 2048x2048 matrix. Class equality via WMMA one-hot GEMM (16x16x32 f16,
// K padded to 32 >= 20 classes): D[m][n] = sum_k 1[cls_m==k]*1[cls_n==k].
// Operands are synthesized directly in the ISA register layout (§7.12.2).
// IoU threshold is branch-free (inter >= t*den, den > 0), predicates combine
// with bitwise AND, bits pack via wave32 ballot, and the 16 result words are
// distributed to lanes with cndmask selects so the tile is written with a
// single predicated store (lanes 0..15) instead of 16 branchy stores.
// ---------------------------------------------------------------------------
__global__ void __launch_bounds__(32)
mask_kernel(const float4* __restrict__ corners, const int* __restrict__ cls,
            unsigned int* __restrict__ mask)
{
    const int lane  = threadIdx.x;
    const int chunk = blockIdx.x;   // 0..63  : 32-col word
    const int rb    = blockIdx.y;   // 0..127 : 16-row band
    const int b     = blockIdx.z;
    const int i0 = rb * 16;
    const int j0 = chunk * 32;
    unsigned int* mrow = mask + ((size_t)b * TOPK + i0) * MASK_WORDS + chunk;

    if (j0 + 31 <= i0) {            // whole tile has j <= i: no bits (j>i required)
        if (lane < 16) mrow[lane * MASK_WORDS] = 0u;
        return;
    }

    __shared__ float4 rowc[16];
    __shared__ float4 colc[32];
    __shared__ int    rcls[16];
    __shared__ int    ccls[32];
    const float4* cb = corners + (size_t)b * TOPK;
    const int*    kb = cls     + (size_t)b * TOPK;
    colc[lane] = cb[j0 + lane];
    ccls[lane] = kb[j0 + lane];
    if (lane < 16) { rowc[lane] = cb[i0 + lane]; rcls[lane] = kb[i0 + lane]; }
    __syncthreads();

    // Build one-hot operands in WMMA register layout:
    //   lane<16: K = 0..7 (v0-3), 16..23 (v4-7);  lane>=16: K = 8..15, 24..31.
    const int mn    = lane & 15;
    const int kbase = (lane < 16) ? 0 : 8;
    const int cm  = rcls[mn];
    const int cn0 = ccls[mn];
    const int cn1 = ccls[mn + 16];
    v16h A, B0, B1;
    #pragma unroll
    for (int h = 0; h < 16; ++h) {
        int K = (h < 8) ? (kbase + h) : (kbase + 8 + h);   // K in [0,32)
        A[h]  = (cm  == K) ? (_Float16)1.0f : (_Float16)0.0f;
        B0[h] = (cn0 == K) ? (_Float16)1.0f : (_Float16)0.0f;
        B1[h] = (cn1 == K) ? (_Float16)1.0f : (_Float16)0.0f;
    }
    v8f zero = {};
    v8f eq0 = __builtin_amdgcn_wmma_f32_16x16x32_f16(false, A, false, B0, (short)0, zero, false, false);
    v8f eq1 = __builtin_amdgcn_wmma_f32_16x16x32_f16(false, A, false, B1, (short)0, zero, false, false);

    // C/D layout: lane<16 -> M=r, lane>=16 -> M=r+8; N = lane&15.
    const int mofs = (lane < 16) ? 0 : 8;
    const int n    = lane & 15;
    const float4 c0 = colc[n];
    const float4 c1 = colc[n + 16];
    const float areaC0 = (c0.z - c0.x + 1.0f) * (c0.w - c0.y + 1.0f);
    const float areaC1 = (c1.z - c1.x + 1.0f) * (c1.w - c1.y + 1.0f);

    unsigned int myword = 0u;       // word this lane will store (lanes 0..15)
    #pragma unroll
    for (int r = 0; r < 8; ++r) {
        const int m = r + mofs;
        const float4 rc = rowc[m];
        const int gi = i0 + m;
        const float areaR = (rc.z - rc.x + 1.0f) * (rc.w - rc.y + 1.0f);

        float iw0 = fmaxf(fminf(rc.z, c0.z) - fmaxf(rc.x, c0.x) + 1.0f, 0.0f);
        float ih0 = fmaxf(fminf(rc.w, c0.w) - fmaxf(rc.y, c0.y) + 1.0f, 0.0f);
        float in0 = iw0 * ih0;
        float iw1 = fmaxf(fminf(rc.z, c1.z) - fmaxf(rc.x, c1.x) + 1.0f, 0.0f);
        float ih1 = fmaxf(fminf(rc.w, c1.w) - fmaxf(rc.y, c1.y) + 1.0f, 0.0f);
        float in1 = iw1 * ih1;
        // iou >= t  <=>  inter >= t * (areaR + areaC - inter + eps)   (den > 0)
        int hit0 = (in0 >= NMS_TH * (areaR + areaC0 - in0 + 1e-16f));
        int hit1 = (in1 >= NMS_TH * (areaR + areaC1 - in1 + 1e-16f));
        int eqA  = (eq0[r] > 0.5f);
        int eqB  = (eq1[r] > 0.5f);
        int jgtA = (j0 + n      > gi);
        int jgtB = (j0 + 16 + n > gi);
        int bA = eqA & hit0 & jgtA;     // bitwise: no EXEC divergence
        int bB = eqB & hit1 & jgtB;

        unsigned int balA = (unsigned int)__ballot(bA);   // low16=row r, high16=row r+8
        unsigned int balB = (unsigned int)__ballot(bB);
        unsigned int wlo = (balA & 0xffffu) | ((balB & 0xffffu) << 16);  // row r
        unsigned int whi = (balA >> 16)     | (balB & 0xffff0000u);      // row r+8
        // Select into this lane's word (ballots are wave-uniform) -> cndmask,
        // then one store at the end instead of 16 branchy stores.
        myword = (lane == r)     ? wlo : myword;
        myword = (lane == r + 8) ? whi : myword;
    }
    if (lane < 16) mrow[lane * MASK_WORDS] = myword;
}

// ---------------------------------------------------------------------------
// Kernel 5: greedy serial scan, one wave per image. Suppression accumulator
// (64 words) lives in 2 VGPRs/lane; bit broadcast via shfl; next mask row is
// software-pipelined to hide L2 latency on the serial chain.
// ---------------------------------------------------------------------------
__global__ void __launch_bounds__(32)
nms_scan_kernel(const float* __restrict__ topconf,
                const unsigned int* __restrict__ mask,
                float* __restrict__ keep)
{
    const int lane = threadIdx.x;
    const int b    = blockIdx.x;
    const unsigned int* mb = mask + (size_t)b * TOPK * MASK_WORDS;
    const float* cf = topconf + b * TOPK;
    float* kp = keep + b * TOPK;

    unsigned int s0 = 0u, s1 = 0u;                 // words [lane], [lane+32]
    unsigned int cur0 = mb[lane], cur1 = mb[lane + 32];
    for (int i = 0; i < TOPK; ++i) {
        unsigned int nx0 = 0u, nx1 = 0u;
        if (i + 1 < TOPK) {                        // prefetch next row off the chain
            const unsigned int* nr = mb + (size_t)(i + 1) * MASK_WORDS;
            nx0 = nr[lane]; nx1 = nr[lane + 32];
        }
        int w = i >> 5;
        unsigned int word = __shfl((w < 32) ? s0 : s1, w & 31, 32);
        int suppressed = (word >> (i & 31)) & 1u;
        int ki = (!suppressed) & (cf[i] >= CONF_TH);
        if (ki) { s0 |= cur0; s1 |= cur1; }
        if (lane == 0) kp[i] = ki ? 1.0f : 0.0f;
        cur0 = nx0; cur1 = nx1;
    }
}

// ---------------------------------------------------------------------------
extern "C" void kernel_launch(void* const* d_in, const int* in_sizes, int n_in,
                              void* d_out, int out_size, void* d_ws, size_t ws_size,
                              hipStream_t stream)
{
    (void)in_sizes; (void)n_in; (void)out_size; (void)ws_size;
    const float* in      = (const float*)d_in[0];
    const float* anchors = (const float*)d_in[1];
    float* out = (float*)d_out;
    char*  ws  = (char*)d_ws;

    float*        conf_ws = (float*)(ws + WS_CONF);
    float*        topconf = (float*)(ws + WS_TOPCONF);
    int*          topidx  = (int*)(ws + WS_TOPIDX);
    float4*       corners = (float4*)(ws + WS_CORN);
    int*          clsbuf  = (int*)(ws + WS_CLS);
    unsigned int* maskbuf = (unsigned int*)(ws + WS_MASK);

    float* out_decode = out + OUT_DECODE;
    int*   out_topidx = (int*)(out + OUT_TOPIDX);
    float* out_keep   = out + OUT_KEEP;

    decode_kernel<<<dim3((HW + 255) / 256, NUM_A, NUM_B), 256, 0, stream>>>(
        in, anchors, out_decode, conf_ws);
    topk_kernel<<<NUM_B, 1024, 0, stream>>>(conf_ws, topconf, topidx, out_topidx);
    gather_kernel<<<(NUM_B * TOPK + 255) / 256, 256, 0, stream>>>(
        out_decode, topidx, corners, clsbuf);
    mask_kernel<<<dim3(MASK_WORDS, TOPK / 16, NUM_B), 32, 0, stream>>>(
        corners, clsbuf, maskbuf);
    nms_scan_kernel<<<NUM_B, 32, 0, stream>>>(topconf, maskbuf, out_keep);
}
